// Head_17875653886334
// MI455X (gfx1250) — matrix operand order repaired and checked by reading, not compiled, not run on hardware
//
#include <hip/hip_runtime.h>
#include <math.h>

// MI455X / gfx1250: wave32, WMMA bf16 16x16x32 (f32 acc).
// Kernel 1: fused QKV projection GEMM [128 x 600] @ [600 x 3*80] per batch,
//           full transposed-weight matrix resident in LDS (285 KB of 320 KB).
// Kernel 2: fused causal attention per batch, async global->LDS staging.

typedef __bf16 v16bf __attribute__((ext_vector_type(16)));
typedef float  v8f   __attribute__((ext_vector_type(8)));

union Frag  { v16bf v; uint4 u[2]; };
union Pack8 { __bf16 h[8]; uint4 u; };

static __device__ __forceinline__ __bf16 f2bf(float f) { return (__bf16)f; }

#define B_   512
#define T_   128
#define C_   600
#define H_   75
#define HP   96      // per-matrix padded head size in workspace
#define NQK  192     // qk row width: q(96) | k(96)
#define KCH  19      // ceil(608/32) K chunks

// ---------------------------------------------------------------------------
// Kernel 1
// ---------------------------------------------------------------------------
__global__ __launch_bounds__(256)
void proj_kernel(const float* __restrict__ x,
                 const float* __restrict__ Wq,
                 const float* __restrict__ Wk,
                 const float* __restrict__ Wv,
                 __bf16* __restrict__ qk,
                 __bf16* __restrict__ vt)
{
    // Full Wt, chunk-major: wt[kb][n=240][k=32] bf16 = 291,840 B
    __shared__ __align__(16) __bf16 wt[KCH * 240 * 32];

    const int b    = blockIdx.x;
    const int tid  = threadIdx.x;
    const int wave = tid >> 5;
    const int lane = tid & 31;
    const int half = lane >> 4;
    const int lm   = lane & 15;

    // Zero the H padding (h = 80..95) in the workspace. h = 75..79 become
    // exact zeros from the zero-padded Wt columns.
    for (int idx = tid; idx < 4096; idx += 256) {
        int row = idx >> 5, c = idx & 31, mat = c >> 4, hh = 80 + (c & 15);
        qk[((size_t)b * T_ + row) * NQK + mat * HP + hh] = f2bf(0.f);
    }
    for (int idx = tid; idx < 2048; idx += 256) {
        int h = 80 + (idx >> 7), t = idx & 127;
        vt[(size_t)b * HP * T_ + (size_t)h * T_ + t] = f2bf(0.f);
    }

    // Stage ALL transposed weights once: wt[k/32][mat*80+h][k%32].
    // 8 k-values packed per b128 LDS store; global reads coalesced over h.
    #pragma unroll 2
    for (int i = 0; i < 72; ++i) {
        int idx = i * 256 + tid;                  // 240 rows * 76 k-groups
        if (idx < 18240) {
            int row = idx % 240;
            int kg  = idx / 240;                  // 0..75, 8 k's each
            int mat = row / 80, h = row % 80;
            int kbase = kg * 8;
            const float* W = (mat == 0) ? Wq : (mat == 1) ? Wk : Wv;
            Pack8 p;
            #pragma unroll
            for (int j = 0; j < 8; ++j) {
                int k = kbase + j;
                float w = (h < H_ && k < C_) ? W[(size_t)k * H_ + h] : 0.f;
                p.h[j] = f2bf(w);
            }
            int chunk = kbase >> 5, col = kbase & 31;
            *(uint4*)&wt[chunk * 7680 + row * 32 + col] = p.u;
        }
    }
    __syncthreads();

    v8f acc[15] = {};                             // 15 N-tiles (3 mats x 80)

    const int trow = 16 * wave + lm;              // A-fragment M row
    const float* xrow = x + ((size_t)b * T_ + trow) * C_;

    // A-fragment loader (contiguous K order: per lane 16 contiguous K
    // starting at 16 * (lane/16)).
    auto loadA = [&](int kb, float* f) {
        const int k0  = kb * 32;
        const int ks1 = k0 + 16 * half;
        const int ks2 = k0 + 16 * half + 8;
        if (k0 + 31 < C_) {
            float4 c0 = *(const float4*)(xrow + ks1);
            float4 c1 = *(const float4*)(xrow + ks1 + 4);
            float4 c2 = *(const float4*)(xrow + ks2);
            float4 c3 = *(const float4*)(xrow + ks2 + 4);
            f[0]=c0.x;  f[1]=c0.y;  f[2]=c0.z;  f[3]=c0.w;
            f[4]=c1.x;  f[5]=c1.y;  f[6]=c1.z;  f[7]=c1.w;
            f[8]=c2.x;  f[9]=c2.y;  f[10]=c2.z; f[11]=c2.w;
            f[12]=c3.x; f[13]=c3.y; f[14]=c3.z; f[15]=c3.w;
        } else {
            #pragma unroll
            for (int j = 0; j < 8; ++j) f[j]     = (ks1 + j < C_) ? xrow[ks1 + j] : 0.f;
            #pragma unroll
            for (int j = 0; j < 8; ++j) f[8 + j] = (ks2 + j < C_) ? xrow[ks2 + j] : 0.f;
        }
    };

    float fs[16];
    loadA(0, fs);

    for (int kb = 0; kb < KCH; ++kb) {
        // Software pipeline: issue next A-tile global loads first.
        float fsN[16];
        if (kb + 1 < KCH) {
            loadA(kb + 1, fsN);
            __builtin_prefetch(xrow + (kb + 2) * 32, 0, 0);
        } else {
            #pragma unroll
            for (int j = 0; j < 16; ++j) fsN[j] = 0.f;
        }

        Frag a;
        #pragma unroll
        for (int j = 0; j < 16; ++j) a.v[j] = f2bf(fs[j]);

        const __bf16* wchunk = wt + kb * 7680;

        // Double-buffered B fragments: load jt+1 before WMMA jt.
        Frag bb[2];
        {
            const uint4* p = (const uint4*)&wchunk[lm * 32 + 16 * half];
            bb[0].u[0] = p[0]; bb[0].u[1] = p[1];
        }
        #pragma unroll
        for (int jt = 0; jt < 15; ++jt) {
            if (jt + 1 < 15) {
                const uint4* p =
                    (const uint4*)&wchunk[(16 * (jt + 1) + lm) * 32 + 16 * half];
                bb[(jt + 1) & 1].u[0] = p[0];
                bb[(jt + 1) & 1].u[1] = p[1];
            }
            acc[jt] = __builtin_amdgcn_wmma_f32_16x16x32_bf16(
                false, a.v, false, bb[jt & 1].v, (short)0, acc[jt], false, false);
        }

        #pragma unroll
        for (int j = 0; j < 16; ++j) fs[j] = fsN[j];
    }

    // Epilogue. q/k: scattered bf16 stores. v: transposed, so the 8 C-rows a
    // lane holds are contiguous in vt -> single b128 store per tile.
    #pragma unroll
    for (int jt = 0; jt < 10; ++jt) {
        int mat = jt / 5;
        int h   = (jt % 5) * 16 + lm;
        #pragma unroll
        for (int g = 0; g < 8; ++g) {
            int row = 16 * wave + 8 * half + g;
            qk[((size_t)b * T_ + row) * NQK + mat * HP + h] = f2bf(acc[jt][g]);
        }
    }
    #pragma unroll
    for (int jt = 10; jt < 15; ++jt) {
        int h = (jt - 10) * 16 + lm;
        Pack8 p;
        #pragma unroll
        for (int g = 0; g < 8; ++g) p.h[g] = f2bf(acc[jt][g]);
        *(uint4*)&vt[(size_t)b * HP * T_ + (size_t)h * T_ + 16 * wave + 8 * half] = p.u;
    }
}

// ---------------------------------------------------------------------------
// Kernel 2: per-batch causal attention. Each wave owns 16 query rows.
// ---------------------------------------------------------------------------
__global__ __launch_bounds__(256)
void attn_kernel(const __bf16* __restrict__ qk,
                 const __bf16* __restrict__ vt,
                 float* __restrict__ out)
{
    __shared__ __align__(16) __bf16 kl[128 * 96];    // K rows      (24 KB)
    __shared__ __align__(16) __bf16 vtl[96 * 128];   // V^T rows    (24 KB)
    __shared__ __align__(16) __bf16 pl[128 * 136];   // P (padded)  (34 KB)

    const int b    = blockIdx.x;
    const int tid  = threadIdx.x;
    const int wave = tid >> 5;
    const int lane = tid & 31;
    const int half = lane >> 4;
    const int lm   = lane & 15;
    const int trow = 16 * wave + lm;

    // Async global->LDS staging (CDNA5 ASYNCcnt path): per-lane b128 copies.
    #pragma unroll
    for (int i = 0; i < 6; ++i) {
        int idx = i * 256 + tid;                     // 128 rows * 12 uint4
        int t = idx / 12, c = idx % 12;
        unsigned dsoff = (unsigned)(uintptr_t)&kl[t * 96 + c * 8];
        const __bf16* g = qk + ((size_t)b * T_ + t) * NQK + HP + c * 8;
        asm volatile("global_load_async_to_lds_b128 %0, %1, off"
                     :: "v"(dsoff), "v"((unsigned long long)(uintptr_t)g)
                     : "memory");
    }
    #pragma unroll
    for (int i = 0; i < 6; ++i) {
        int idx = i * 256 + tid;                     // 96*128 bf16 = 1536 u4
        unsigned dsoff = (unsigned)(uintptr_t)&vtl[idx * 8];
        const __bf16* g = vt + (size_t)b * HP * T_ + idx * 8;
        asm volatile("global_load_async_to_lds_b128 %0, %1, off"
                     :: "v"(dsoff), "v"((unsigned long long)(uintptr_t)g)
                     : "memory");
    }
    asm volatile("s_wait_asynccnt 0x0" ::: "memory");
    __syncthreads();

    // S = Q @ K^T  (K-dim = 96, zero padded above 75)
    v8f sacc[8] = {};
    const __bf16* qrow = qk + ((size_t)b * T_ + trow) * NQK;   // q at offset 0
    #pragma unroll
    for (int kc = 0; kc < 3; ++kc) {
        int k0 = 32 * kc;
        Frag a;
        a.u[0] = *(const uint4*)(qrow + k0 + 16 * half);
        a.u[1] = *(const uint4*)(qrow + k0 + 16 * half + 8);
        Frag bb[2];
        {
            const uint4* p = (const uint4*)&kl[lm * 96 + k0 + 16 * half];
            bb[0].u[0] = p[0]; bb[0].u[1] = p[1];
        }
        #pragma unroll
        for (int j = 0; j < 8; ++j) {
            if (j + 1 < 8) {
                const uint4* p =
                    (const uint4*)&kl[(16 * (j + 1) + lm) * 96 + k0 + 16 * half];
                bb[(j + 1) & 1].u[0] = p[0];
                bb[(j + 1) & 1].u[1] = p[1];
            }
            sacc[j] = __builtin_amdgcn_wmma_f32_16x16x32_bf16(
                false, a.v, false, bb[j & 1].v, (short)0, sacc[j], false, false);
        }
    }

    // Scale + causal mask + softmax, fully in C-fragment layout.
    const float scale = 0.04082482904638630164f;     // 600^-0.5 (C = n_embd)
    const float NEG = -__builtin_inff();
    float rmax[8], rsum[8];
    #pragma unroll
    for (int g = 0; g < 8; ++g) rmax[g] = NEG;
    #pragma unroll
    for (int j = 0; j < 8; ++j) {
        int s = 16 * j + lm;
        #pragma unroll
        for (int g = 0; g < 8; ++g) {
            int m = 16 * wave + 8 * half + g;
            float v = sacc[j][g] * scale;
            v = (s <= m) ? v : NEG;
            sacc[j][g] = v;
            rmax[g] = fmaxf(rmax[g], v);
        }
    }
    #pragma unroll
    for (int g = 0; g < 8; ++g) {
        #pragma unroll
        for (int off = 8; off >= 1; off >>= 1)
            rmax[g] = fmaxf(rmax[g], __shfl_xor(rmax[g], off, 32));
        rsum[g] = 0.f;
    }
    #pragma unroll
    for (int j = 0; j < 8; ++j) {
        #pragma unroll
        for (int g = 0; g < 8; ++g) {
            float e = __expf(sacc[j][g] - rmax[g]);  // exp(-inf)=0 when masked
            sacc[j][g] = e;
            rsum[g] += e;
        }
    }
    #pragma unroll
    for (int g = 0; g < 8; ++g) {
        #pragma unroll
        for (int off = 8; off >= 1; off >>= 1)
            rsum[g] += __shfl_xor(rsum[g], off, 32);
    }
    // Re-layout P: C-fragment -> LDS -> A-fragment.
    #pragma unroll
    for (int j = 0; j < 8; ++j) {
        #pragma unroll
        for (int g = 0; g < 8; ++g) {
            int m = 16 * wave + 8 * half + g, s = 16 * j + lm;
            pl[m * 136 + s] = f2bf(sacc[j][g] / rsum[g]);
        }
    }
    __syncthreads();

    // O = P @ V  (K-dim = 128; B fragments contiguous thanks to V^T)
    v8f oacc[6] = {};
    const __bf16* prow = pl + trow * 136;
    #pragma unroll
    for (int kc = 0; kc < 4; ++kc) {
        int k0 = 32 * kc;
        Frag a;
        a.u[0] = *(const uint4*)(prow + k0 + 16 * half);
        a.u[1] = *(const uint4*)(prow + k0 + 16 * half + 8);
        Frag bb[2];
        {
            const uint4* p = (const uint4*)&vtl[lm * 128 + k0 + 16 * half];
            bb[0].u[0] = p[0]; bb[0].u[1] = p[1];
        }
        #pragma unroll
        for (int jh = 0; jh < 6; ++jh) {
            if (jh + 1 < 6) {
                const uint4* p =
                    (const uint4*)&vtl[(16 * (jh + 1) + lm) * 128 + k0 + 16 * half];
                bb[(jh + 1) & 1].u[0] = p[0];
                bb[(jh + 1) & 1].u[1] = p[1];
            }
            oacc[jh] = __builtin_amdgcn_wmma_f32_16x16x32_bf16(
                false, a.v, false, bb[jh & 1].v, (short)0, oacc[jh], false, false);
        }
    }

    // Store f32 output, only h < 75.
    #pragma unroll
    for (int jh = 0; jh < 6; ++jh) {
        int h = 16 * jh + lm;
        if (h < H_) {
            #pragma unroll
            for (int g = 0; g < 8; ++g) {
                int t = 16 * wave + 8 * half + g;
                out[((size_t)b * T_ + t) * H_ + h] = oacc[jh][g];
            }
        }
    }
}

// ---------------------------------------------------------------------------
extern "C" void kernel_launch(void* const* d_in, const int* in_sizes, int n_in,
                              void* d_out, int out_size, void* d_ws, size_t ws_size,
                              hipStream_t stream) {
    const float* x  = (const float*)d_in[0];
    const float* Wq = (const float*)d_in[1];
    const float* Wk = (const float*)d_in[2];
    const float* Wv = (const float*)d_in[3];

    // Workspace: qk bf16 [B][T][192] then vt bf16 [B][96][128]  (~36 MB)
    __bf16* qk = (__bf16*)d_ws;
    __bf16* vt = qk + (size_t)B_ * T_ * NQK;

    proj_kernel<<<B_, 256, 0, stream>>>(x, Wq, Wk, Wv, qk, vt);
    attn_kernel<<<B_, 256, 0, stream>>>(qk, vt, (float*)d_out);
}
